// LongformerSelfAttention_18107582120751
// MI455X (gfx1250) — matrix operand order repaired and checked by reading, not compiled, not run on hardware
//
#include <hip/hip_runtime.h>
#include <hip/hip_fp16.h>
#include <cstdint>

typedef __attribute__((ext_vector_type(16))) _Float16 v16h;
typedef __attribute__((ext_vector_type(8)))  _Float16 v8h;
typedef __attribute__((ext_vector_type(8)))  float    v8f;

#define BB 2
#define SS 4096
#define DM 768
#define NHEAD 12
#define HDIM 64
#define WIN 256
#define GG 16
#define NCOL 544            /* 16 sel cols + 528 band-window cols */
#define NTILE 34            /* 1 sel tile + 33 band tiles of 16 keys */
#define SPAD (SS + 2 * WIN) /* padded key count (K rows / V columns) */
#define VTP SPAD            /* padded key pitch of transposed V */

__device__ __forceinline__ v8f wmma16(v16h a, v16h b, v8f c) {
  return __builtin_amdgcn_wmma_f32_16x16x32_f16(false, a, false, b, (short)0, c,
                                                false, false);
}

// Async global->LDS 16-byte copy (CDNA5 GLOBAL_LOAD_ASYNC_TO_LDS_B128).
// LDS destination address = low 32 bits of the generic shared pointer
// (LDS aperture: addr[31:0] is the in-wave LDS offset).
__device__ __forceinline__ void async_b128(void* lds, const void* gaddr) {
  asm volatile("global_load_async_to_lds_b128 %0, %1, off"
               :
               : "v"((unsigned)(uintptr_t)lds), "v"(gaddr)
               : "memory");
}
__device__ __forceinline__ void wait_async0() {
  asm volatile("s_wait_asynccnt 0x0" ::: "memory");
}
__device__ __forceinline__ void wait_ds0() {
  asm volatile("s_wait_dscnt 0x0" ::: "memory");
}

// A fragment (16x32 f16, MxK) from row-major LDS tile, leading dim ld (halfs).
__device__ __forceinline__ v16h load_a_frag(const _Float16* p, int ld, int lane) {
  int m  = lane & 15;
  int kb = (lane >> 4) * 8;
  const _Float16* row = p + m * ld + kb;
  v8h lo = *(const v8h*)(row);
  v8h hi = *(const v8h*)(row + 16);
  return __builtin_shufflevector(lo, hi, 0, 1, 2, 3, 4, 5, 6, 7, 8, 9, 10, 11,
                                 12, 13, 14, 15);
}

// A fragment from an f32 tile (probabilities), converted to f16. Vector reads.
__device__ __forceinline__ v16h load_a_frag_f32(const float* p, int ld, int lane) {
  int m  = lane & 15;
  int kb = (lane >> 4) * 8;
  const float* row = p + m * ld + kb;
  float4 x0 = ((const float4*)row)[0];
  float4 x1 = ((const float4*)row)[1];
  float4 x2 = ((const float4*)(row + 16))[0];
  float4 x3 = ((const float4*)(row + 16))[1];
  v16h a;
  a[0]  = (_Float16)x0.x; a[1]  = (_Float16)x0.y;
  a[2]  = (_Float16)x0.z; a[3]  = (_Float16)x0.w;
  a[4]  = (_Float16)x1.x; a[5]  = (_Float16)x1.y;
  a[6]  = (_Float16)x1.z; a[7]  = (_Float16)x1.w;
  a[8]  = (_Float16)x2.x; a[9]  = (_Float16)x2.y;
  a[10] = (_Float16)x2.z; a[11] = (_Float16)x2.w;
  a[12] = (_Float16)x3.x; a[13] = (_Float16)x3.y;
  a[14] = (_Float16)x3.z; a[15] = (_Float16)x3.w;
  return a;
}

// B fragment (KxN = 32x16) from an N-major tile: p[n*ld + k]. Per lane this is
// two 8-half runs at k = 8h and k = 16 + 8h -> two ds_load_b128.
__device__ __forceinline__ v16h load_bT_frag(const _Float16* p, int ld, int lane) {
  int n  = lane & 15;
  int kb = (lane >> 4) * 8;
  const _Float16* col = p + n * ld + kb;
  v8h lo = *(const v8h*)(col);
  v8h hi = *(const v8h*)(col + 16);
  return __builtin_shufflevector(lo, hi, 0, 1, 2, 3, 4, 5, 6, 7, 8, 9, 10, 11,
                                 12, 13, 14, 15);
}

// ---------------------------------------------------------------------------
// One-time weight transpose: WT(f16)[n*768 + k] = (f16) W(f32)[k*768 + n].
// ---------------------------------------------------------------------------
__global__ __launch_bounds__(128) void transpose_w(const float* __restrict__ W,
                                                   _Float16* __restrict__ WT) {
  __shared__ alignas(32) _Float16 sT[64 * 80];
  const int tid = threadIdx.x;
  const int k0 = blockIdx.x * 64, n0 = blockIdx.y * 64;
  {
    int r = tid >> 1, cs = (tid & 1) * 32;  // k-local row, 32 n's
    const float4* src = (const float4*)(W + (size_t)(k0 + r) * DM + n0 + cs);
#pragma unroll
    for (int i = 0; i < 4; ++i) {
      float4 a = src[2 * i], b = src[2 * i + 1];
      v8h h;
      h[0] = (_Float16)a.x; h[1] = (_Float16)a.y;
      h[2] = (_Float16)a.z; h[3] = (_Float16)a.w;
      h[4] = (_Float16)b.x; h[5] = (_Float16)b.y;
      h[6] = (_Float16)b.z; h[7] = (_Float16)b.w;
      *(v8h*)(sT + r * 80 + cs + 8 * i) = h;
    }
  }
  __syncthreads();
  {
    int n = tid >> 1, ks = (tid & 1) * 32;  // n-local row, 32 k's
    _Float16* dst = WT + (size_t)(n0 + n) * DM + k0 + ks;
#pragma unroll
    for (int j = 0; j < 4; ++j) {
      v8h h;
#pragma unroll
      for (int i = 0; i < 8; ++i) h[i] = sT[(ks + 8 * j + i) * 80 + n];
      *(v8h*)(dst + 8 * j) = h;
    }
  }
}

// ---------------------------------------------------------------------------
// Projection GEMM: (X(f32) @ W + bias) * scale, using WT (f16, N-major).
// Writes Y with per-batch row pitch/pad (for padded K), and/or a per-(b,h)
// transposed copy YT[(b,h,col)*ytPitch + ytPad + s].
// ---------------------------------------------------------------------------
__global__ __launch_bounds__(128) void proj_gemm(
    const float* __restrict__ X, const _Float16* __restrict__ WT,
    const float* __restrict__ bias, _Float16* __restrict__ Y,
    int yRowPitch, int yRowPad,
    _Float16* __restrict__ YT, int ytPitch, int ytPad,
    int M, float scale, int rowmap) {
  __shared__ alignas(32) _Float16 sA[64 * 32];   // rows x k
  __shared__ alignas(32) _Float16 sBt[64 * 32];  // n-major: n x k
  const int tid = threadIdx.x, lane = tid & 31, wid = tid >> 5;
  const int m0 = blockIdx.x * 64, n0 = blockIdx.y * 64;

  const int ar = tid >> 1, ac = (tid & 1) * 16;  // A stage: 64x32
  const int bn = tid >> 1, bs = (tid & 1) * 16;  // B stage: 64(n) x 32(k)
  int mrow = m0 + ar;
  int xrow = rowmap ? ((mrow / rowmap) * SS + (mrow % rowmap)) : mrow;
  bool mok = (mrow < M);

  v8f acc[4] = {};
  for (int k0 = 0; k0 < DM; k0 += 32) {
    {  // B tile: async f16 copy from WT (already N-major)
      const _Float16* wsrc = WT + (size_t)(n0 + bn) * DM + k0 + bs;
      _Float16* wdst = sBt + bn * 32 + bs;
      async_b128(wdst, wsrc);
      async_b128(wdst + 8, wsrc + 8);
    }
    if (mok) {
      const float4* src = (const float4*)(X + (size_t)xrow * DM + k0 + ac);
      __builtin_prefetch(X + (size_t)xrow * DM + k0 + 32, 0, 0);
#pragma unroll
      for (int i = 0; i < 2; ++i) {
        float4 a = src[2 * i], b = src[2 * i + 1];
        v8h h;
        h[0] = (_Float16)a.x; h[1] = (_Float16)a.y;
        h[2] = (_Float16)a.z; h[3] = (_Float16)a.w;
        h[4] = (_Float16)b.x; h[5] = (_Float16)b.y;
        h[6] = (_Float16)b.z; h[7] = (_Float16)b.w;
        *(v8h*)(sA + ar * 32 + ac + 8 * i) = h;
      }
    } else {
      v8h z = {};
      *(v8h*)(sA + ar * 32 + ac + 0) = z;
      *(v8h*)(sA + ar * 32 + ac + 8) = z;
    }
    wait_async0();
    __syncthreads();
    v16h a = load_a_frag(sA + wid * 16 * 32, 32, lane);
#pragma unroll
    for (int nc = 0; nc < 4; ++nc) {
      v16h b = load_bT_frag(sBt + nc * 16 * 32, 32, lane);
      acc[nc] = wmma16(a, b, acc[nc]);
    }
    __syncthreads();
  }
  const int n = lane & 15, hi = lane >> 4;
#pragma unroll
  for (int nc = 0; nc < 4; ++nc)
#pragma unroll
    for (int r = 0; r < 8; ++r) {
      int row = m0 + wid * 16 + r + hi * 8;
      int col = n0 + nc * 16 + n;
      if (row < M) {
        float v = (acc[nc][r] + bias[col]) * scale;
        int bb = row >> 12, s = row & (SS - 1);
        int hh = col >> 6, hc = col & 63;
        if (Y)
          Y[((size_t)bb * yRowPitch + yRowPad + s) * DM + col] = (_Float16)v;
        if (YT)
          YT[(((size_t)bb * NHEAD + hh) * HDIM + hc) * (size_t)ytPitch + ytPad +
             s] = (_Float16)v;
      }
    }
}

// ---------------------------------------------------------------------------
// Banded attention + global-key (sel) columns.
// One block (4 waves) per (b, h, 16-query tile). Scores (544 cols) in LDS,
// exact softmax, PV with unnormalized exp. K and V come from zero-padded
// layouts -> fully branch-free async b128 staging.
// ---------------------------------------------------------------------------
__global__ __launch_bounds__(128) void band_attn(
    const _Float16* __restrict__ Qm, const _Float16* __restrict__ Kpad,
    const _Float16* __restrict__ VT, float* __restrict__ out) {
  __shared__ alignas(32) _Float16 sQ[16 * 64];
  __shared__ alignas(32) _Float16 sK[4][16 * 64];
  __shared__ alignas(32) _Float16 sVt[64 * 32];  // n-major: hdcol x key
  __shared__ float sS[16 * NCOL];
  __shared__ float sRed[16][8];
  __shared__ float sMax[16];
  __shared__ float sSum[16];
  const int tid = threadIdx.x, lane = tid & 31, wid = tid >> 5;
  const int h = blockIdx.y, b = blockIdx.z;
  const int qb = blockIdx.x * 16;

  {  // stage Q tile (16x64 f16), async copy
    int r = tid >> 3, c = (tid & 7) * 8;
    async_b128(sQ + r * 64 + c,
               Qm + ((size_t)b * SS + qb + r) * DM + h * HDIM + c);
    wait_async0();
  }
  __syncthreads();
  v16h a0 = load_a_frag(sQ, 64, lane);       // hd 0..31
  v16h a1 = load_a_frag(sQ + 32, 64, lane);  // hd 32..63

  // Phase 1: scores -> sS, wave-strided over 34 key tiles (branch-free stage).
  for (int t = wid; t < NTILE; t += 4) {
    int krow = lane & 15, seg = lane >> 4;
    int kpos = (t == 0) ? krow : (qb - WIN + (t - 1) * 16 + krow);
    const _Float16* src =
        Kpad + ((size_t)b * SPAD + WIN + kpos) * DM + h * HDIM + seg * 32;
    _Float16* dst = sK[wid] + krow * 64 + seg * 32;
#pragma unroll
    for (int i = 0; i < 4; ++i) async_b128(dst + 8 * i, src + 8 * i);
    wait_async0();
    v16h b0 = load_bT_frag(sK[wid], 64, lane);
    v16h b1 = load_bT_frag(sK[wid] + 32, 64, lane);
    v8f c = {};
    c = wmma16(a0, b0, c);
    c = wmma16(a1, b1, c);
    int n = lane & 15, hi = lane >> 4;
#pragma unroll
    for (int r = 0; r < 8; ++r) {
      int m = r + hi * 8;
      int col = t * 16 + n;
      float v = c[r];
      if (t == 0) {            // sel columns (global keys 0..g-1)
        if (n >= GG) v = -1e30f;
      } else {                 // band columns
        int cb = col - 16;
        int kp = qb - WIN + cb;
        bool valid = (cb >= m) && (cb <= m + 2 * WIN) && (kp >= 0) && (kp < SS);
        v = valid ? (v + ((kp < GG) ? -10000.0f : 0.0f)) : -1e30f;
      }
      sS[m * NCOL + col] = v;
    }
  }
  __syncthreads();

  // Phase 2: softmax (exp(s-max) kept unnormalized in sS, sums in sSum)
  {
    int row = tid & 15, part = tid >> 4;
    float pm = -1e30f;
    for (int c = part; c < NCOL; c += 8) pm = fmaxf(pm, sS[row * NCOL + c]);
    sRed[row][part] = pm;
    __syncthreads();
    if (tid < 16) {
      float m = sRed[tid][0];
#pragma unroll
      for (int i = 1; i < 8; ++i) m = fmaxf(m, sRed[tid][i]);
      sMax[tid] = m;
    }
    __syncthreads();
    float rm = sMax[row], ps = 0.f;
    for (int c = part; c < NCOL; c += 8) {
      float e = __expf(sS[row * NCOL + c] - rm);
      sS[row * NCOL + c] = e;
      ps += e;
    }
    sRed[row][part] = ps;
    __syncthreads();
    if (tid < 16) {
      float s = 0.f;
#pragma unroll
      for (int i = 0; i < 8; ++i) s += sRed[tid][i];
      sSum[tid] = s;
    }
  }

  // Phase 3: O = P x V over 17 chunks of 32 columns; wave owns 16 output cols.
  v8f acc = {};
  for (int kt = 0; kt < NCOL / 32; ++kt) {
    __syncthreads();
    {  // branch-free async stage of V^T: 64 hd-cols x 32 keys
      int nloc = tid >> 1;
      int halfsel = tid & 1;
      int cbase = kt * 32 + halfsel * 16;
      int keyBase = cbase ? (qb - WIN + cbase - 16) : 0;  // sel keys at cbase==0
      const _Float16* src =
          VT + (((size_t)b * NHEAD + h) * HDIM + nloc) * VTP + WIN + keyBase;
      _Float16* dst = sVt + nloc * 32 + halfsel * 16;
      async_b128(dst, src);
      async_b128(dst + 8, src + 8);
      wait_async0();
    }
    __syncthreads();
    v16h ap = load_a_frag_f32(sS + kt * 32, NCOL, lane);
    v16h bv = load_bT_frag(sVt + wid * 16 * 32, 32, lane);
    acc = wmma16(ap, bv, acc);
  }
  int n = lane & 15, hi = lane >> 4;
#pragma unroll
  for (int r = 0; r < 8; ++r) {
    int m = r + hi * 8;
    out[((size_t)b * SS + qb + m) * DM + h * HDIM + wid * 16 + n] =
        acc[r] * (1.f / sSum[m]);
  }
}

// ---------------------------------------------------------------------------
// Global-token attention: rows 0..g-1 get softmax(qg . kg over all S) @ vg.
// ---------------------------------------------------------------------------
__global__ __launch_bounds__(128) void global_attn(
    const _Float16* __restrict__ QGm, const _Float16* __restrict__ KGm,
    const _Float16* __restrict__ VGT, float* __restrict__ out) {
  __shared__ alignas(32) _Float16 sQ[16 * 64];
  __shared__ alignas(32) _Float16 sK[4][32 * 64];
  __shared__ alignas(32) _Float16 sVt[4][64 * 32];  // per wave: hdcol x key
  __shared__ float sP[4][16 * 32];
  __shared__ float sRedM[4][16][16];
  __shared__ float sMax[16];
  __shared__ float sSum[16];
  __shared__ float sO[16 * 64];
  const int tid = threadIdx.x, lane = tid & 31, wid = tid >> 5;
  const int h = blockIdx.x, b = blockIdx.y;
  const int n = lane & 15, hi = lane >> 4;

  {
    int r = tid >> 3, c = (tid & 7) * 8;
    async_b128(sQ + r * 64 + c,
               QGm + ((size_t)b * GG + r) * DM + h * HDIM + c);
    wait_async0();
  }
  for (int i = tid; i < 16 * 64; i += 128) sO[i] = 0.f;
  __syncthreads();
  v16h a0 = load_a_frag(sQ, 64, lane);
  v16h a1 = load_a_frag(sQ + 32, 64, lane);

  // Pass A: row maxima
  float pmx[8];
#pragma unroll
  for (int r = 0; r < 8; ++r) pmx[r] = -1e30f;
  for (int kt = wid; kt < SS / 32; kt += 4) {
    const _Float16* src = KGm + ((size_t)b * SS + kt * 32 + lane) * DM + h * HDIM;
    _Float16* dst = sK[wid] + lane * 64;
#pragma unroll
    for (int i = 0; i < 8; ++i) async_b128(dst + 8 * i, src + 8 * i);
    wait_async0();
#pragma unroll
    for (int nh = 0; nh < 2; ++nh) {
      v16h b0 = load_bT_frag(sK[wid] + nh * 16 * 64, 64, lane);
      v16h b1 = load_bT_frag(sK[wid] + nh * 16 * 64 + 32, 64, lane);
      v8f c = {};
      c = wmma16(a0, b0, c);
      c = wmma16(a1, b1, c);
#pragma unroll
      for (int r = 0; r < 8; ++r) pmx[r] = fmaxf(pmx[r], c[r]);
    }
  }
#pragma unroll
  for (int r = 0; r < 8; ++r) sRedM[wid][r + hi * 8][n] = pmx[r];
  __syncthreads();
  if (tid < 16) {
    float m = -1e30f;
    for (int w = 0; w < 4; ++w)
      for (int l = 0; l < 16; ++l) m = fmaxf(m, sRedM[w][tid][l]);
    sMax[tid] = m;
  }
  __syncthreads();

  // Pass B: exp, row sums, and PV accumulation
  float psm[8];
#pragma unroll
  for (int r = 0; r < 8; ++r) psm[r] = 0.f;
  v8f accO[4] = {};
  for (int kt = wid; kt < SS / 32; kt += 4) {
    {  // K rows (row-major) + V^T cols (n-major), all async b128 copies
      const _Float16* ksrc =
          KGm + ((size_t)b * SS + kt * 32 + lane) * DM + h * HDIM;
      _Float16* kdst = sK[wid] + lane * 64;
#pragma unroll
      for (int i = 0; i < 8; ++i) async_b128(kdst + 8 * i, ksrc + 8 * i);
#pragma unroll
      for (int rep = 0; rep < 2; ++rep) {
        int nloc = lane + rep * 32;
        const _Float16* vsrc =
            VGT + (((size_t)b * NHEAD + h) * HDIM + nloc) * SS + kt * 32;
        _Float16* vdst = sVt[wid] + nloc * 32;
#pragma unroll
        for (int i = 0; i < 4; ++i) async_b128(vdst + 8 * i, vsrc + 8 * i);
      }
    }
    wait_async0();
#pragma unroll
    for (int nh = 0; nh < 2; ++nh) {
      v16h b0 = load_bT_frag(sK[wid] + nh * 16 * 64, 64, lane);
      v16h b1 = load_bT_frag(sK[wid] + nh * 16 * 64 + 32, 64, lane);
      v8f c = {};
      c = wmma16(a0, b0, c);
      c = wmma16(a1, b1, c);
#pragma unroll
      for (int r = 0; r < 8; ++r) {
        int m = r + hi * 8;
        float e = __expf(c[r] - sMax[m]);
        psm[r] += e;
        sP[wid][m * 32 + nh * 16 + n] = e;
      }
    }
    wait_ds0();
    v16h ap = load_a_frag_f32(sP[wid], 32, lane);
#pragma unroll
    for (int nc = 0; nc < 4; ++nc) {
      v16h bv = load_bT_frag(sVt[wid] + nc * 16 * 32, 32, lane);
      accO[nc] = wmma16(ap, bv, accO[nc]);
    }
  }
#pragma unroll
  for (int r = 0; r < 8; ++r) sRedM[wid][r + hi * 8][n] = psm[r];
  __syncthreads();
  if (tid < 16) {
    float s = 0.f;
    for (int w = 0; w < 4; ++w)
      for (int l = 0; l < 16; ++l) s += sRedM[w][tid][l];
    sSum[tid] = s;
  }
  __syncthreads();
#pragma unroll
  for (int nc = 0; nc < 4; ++nc)
#pragma unroll
    for (int r = 0; r < 8; ++r)
      atomicAdd(&sO[(r + hi * 8) * 64 + nc * 16 + n], accO[nc][r]);
  __syncthreads();
  for (int i = tid; i < 16 * 64; i += 128) {
    int m = i >> 6, c = i & 63;
    out[((size_t)b * SS + m) * DM + h * HDIM + c] = sO[i] / sSum[m];
  }
}

// ---------------------------------------------------------------------------
extern "C" void kernel_launch(void* const* d_in, const int* in_sizes, int n_in,
                              void* d_out, int out_size, void* d_ws,
                              size_t ws_size, hipStream_t stream) {
  const float* hs  = (const float*)d_in[0];
  const float* Wq  = (const float*)d_in[1];
  const float* bq  = (const float*)d_in[2];
  const float* Wk  = (const float*)d_in[3];
  const float* bk  = (const float*)d_in[4];
  const float* Wv  = (const float*)d_in[5];
  const float* bv  = (const float*)d_in[6];
  const float* Wqg = (const float*)d_in[7];
  const float* bqg = (const float*)d_in[8];
  const float* Wkg = (const float*)d_in[9];
  const float* bkg = (const float*)d_in[10];
  const float* Wvg = (const float*)d_in[11];
  const float* bvg = (const float*)d_in[12];
  // d_in[13] = n_global (== 16, fixed by the reference constants)
  float* out = (float*)d_out;

  _Float16* p = (_Float16*)d_ws;
  const size_t per = (size_t)BB * SS * DM;
  const size_t wsz = (size_t)DM * DM;
  const size_t kpadSz = (size_t)BB * SPAD * DM;
  const size_t vtSz   = (size_t)BB * NHEAD * HDIM * VTP;
  _Float16* Qf   = p; p += per;
  _Float16* KGf  = p; p += per;
  _Float16* QGf  = p; p += (size_t)BB * GG * DM;
  _Float16* WTq  = p; p += wsz;
  _Float16* WTk  = p; p += wsz;
  _Float16* WTv  = p; p += wsz;
  _Float16* WTqg = p; p += wsz;
  _Float16* WTkg = p; p += wsz;
  _Float16* WTvg = p; p += wsz;
  _Float16* Kpad = p; p += kpadSz;  // adjacent to VT: one memset covers both
  _Float16* VT   = p; p += vtSz;
  _Float16* VGT  = p; p += (size_t)BB * NHEAD * HDIM * SS;

  const float scale = 0.125f;  // 1/sqrt(64)
  dim3 blk(128);

  // Zero padded K/V buffers (pads must read as 0 on every call).
  hipMemsetAsync(Kpad, 0, (kpadSz + vtSz) * sizeof(_Float16), stream);

  dim3 gt(DM / 64, DM / 64);
  transpose_w<<<gt, blk, 0, stream>>>(Wq, WTq);
  transpose_w<<<gt, blk, 0, stream>>>(Wk, WTk);
  transpose_w<<<gt, blk, 0, stream>>>(Wv, WTv);
  transpose_w<<<gt, blk, 0, stream>>>(Wqg, WTqg);
  transpose_w<<<gt, blk, 0, stream>>>(Wkg, WTkg);
  transpose_w<<<gt, blk, 0, stream>>>(Wvg, WTvg);

  dim3 gp(BB * SS / 64, DM / 64);
  proj_gemm<<<gp, blk, 0, stream>>>(hs, WTq, bq, Qf, SS, 0, nullptr, 0, 0,
                                    BB * SS, scale, 0);
  proj_gemm<<<gp, blk, 0, stream>>>(hs, WTk, bk, Kpad, SPAD, WIN, nullptr, 0, 0,
                                    BB * SS, 1.f, 0);
  proj_gemm<<<gp, blk, 0, stream>>>(hs, WTv, bv, nullptr, 0, 0, VT, VTP, WIN,
                                    BB * SS, 1.f, 0);
  proj_gemm<<<gp, blk, 0, stream>>>(hs, WTkg, bkg, KGf, SS, 0, nullptr, 0, 0,
                                    BB * SS, 1.f, 0);
  proj_gemm<<<gp, blk, 0, stream>>>(hs, WTvg, bvg, nullptr, 0, 0, VGT, SS, 0,
                                    BB * SS, 1.f, 0);
  proj_gemm<<<dim3(1, DM / 64), blk, 0, stream>>>(hs, WTqg, bqg, QGf, SS, 0,
                                                  nullptr, 0, 0, BB * GG, scale,
                                                  GG);

  band_attn<<<dim3(SS / 16, NHEAD, BB), blk, 0, stream>>>(Qf, Kpad, VT, out);
  global_attn<<<dim3(NHEAD, BB), blk, 0, stream>>>(QGf, KGf, VGT, out);
}
